// Net_876173328799
// MI455X (gfx1250) — matrix operand order repaired and checked by reading, not compiled, not run on hardware
//
#include <hip/hip_runtime.h>

// ---------------------------------------------------------------------------
// AGNN forward for MI455X (gfx1250, wave32).
//   GEMMs: v_wmma_f32_16x16x32_bf16 with bf16 hi/lo split (3 WMMAs -> ~f32).
//   Graph attention: wave-per-edge gather kernels; node features (51MB) are
//   L2-resident (192MB), so edge gathers/atomics run at L2 bandwidth.
// ---------------------------------------------------------------------------

typedef __attribute__((ext_vector_type(16))) __bf16 v16bf;
typedef __attribute__((ext_vector_type(8)))  __bf16 v8bf;
typedef __attribute__((ext_vector_type(8)))  float  v8f;

// ---- order-preserving float<->uint encoding for atomic segment-max --------
__device__ __forceinline__ unsigned f2ord(float f) {
  unsigned u = __float_as_uint(f);
  return (u & 0x80000000u) ? ~u : (u | 0x80000000u);
}
__device__ __forceinline__ float ord2f(unsigned u) {
  return __uint_as_float((u & 0x80000000u) ? (u ^ 0x80000000u) : ~u);
}

// ---- f32 -> bf16 (hi) + bf16 (residual lo) --------------------------------
__global__ void k_split_bf16(const float* __restrict__ in,
                             __bf16* __restrict__ hi, __bf16* __restrict__ lo,
                             int n) {
  int i = blockIdx.x * blockDim.x + threadIdx.x;
  if (i >= n) return;
  float v  = in[i];
  __bf16 h = (__bf16)v;
  hi[i]    = h;
  lo[i]    = (__bf16)(v - (float)h);
}

// ---- WMMA GEMM: C[M,Ncol] = A[M,K] * B[Ncol,K]^T + bias, optional relu ----
// Block = 256 threads = 8 waves; block tile = 128 rows x 64 cols;
// each wave: 16 rows x 4 column-tiles of 16.  K stepped by 32.
__global__ void __launch_bounds__(256)
k_gemm_wmma(const __bf16* __restrict__ Ahi, const __bf16* __restrict__ Alo,
            const __bf16* __restrict__ Bhi, const __bf16* __restrict__ Blo,
            const float* __restrict__ bias, float* __restrict__ C,
            int M, int K, int Ncol, int relu) {
  const int lane = threadIdx.x & 31;
  const int wv   = threadIdx.x >> 5;
  const int half = lane >> 4;    // 0: lanes 0-15, 1: lanes 16-31
  const int l15  = lane & 15;
  const int rowBase = blockIdx.x * 128 + wv * 16;
  const int colBase = blockIdx.y * 64;

  // A fragment source row (clamped; stores are guarded below).
  int arow = rowBase + l15;
  if (arow >= M) arow = M - 1;
  // A layout (matches B): lane<16 -> K 0..15; lane>=16 -> K 16..31
  const __bf16* aH = Ahi + (size_t)arow * K + half * 16;
  const __bf16* aL = Alo + (size_t)arow * K + half * 16;

  v8f acc[4];
#pragma unroll
  for (int t = 0; t < 4; ++t) acc[t] = (v8f){0, 0, 0, 0, 0, 0, 0, 0};

  for (int k0 = 0; k0 < K; k0 += 32) {
    v8bf ah0 = *(const v8bf*)(aH + k0);
    v8bf ah1 = *(const v8bf*)(aH + k0 + 8);
    v8bf al0 = *(const v8bf*)(aL + k0);
    v8bf al1 = *(const v8bf*)(aL + k0 + 8);
    v16bf aHi = __builtin_shufflevector(ah0, ah1, 0,1,2,3,4,5,6,7,8,9,10,11,12,13,14,15);
    v16bf aLo = __builtin_shufflevector(al0, al1, 0,1,2,3,4,5,6,7,8,9,10,11,12,13,14,15);
#pragma unroll
    for (int t = 0; t < 4; ++t) {
      // B layout used here, same k order as A (16-bit 32x16): lane holds col N=lane&15, 16 consecutive K
      // (lanes<16: K 0..15; lanes>=16: K 16..31) => one 32B contiguous load.
      int wrow = colBase + t * 16 + l15;                 // row of W = column of B
      const __bf16* bH = Bhi + (size_t)wrow * K + k0 + half * 16;
      const __bf16* bL = Blo + (size_t)wrow * K + k0 + half * 16;
      v16bf bHi = *(const v16bf*)bH;
      v16bf bLo = *(const v16bf*)bL;
      // hi*hi + hi*lo + lo*hi  (bf16 products exact in f32; f32 accumulate)
      acc[t] = __builtin_amdgcn_wmma_f32_16x16x32_bf16(false, aHi, false, bHi, (short)0, acc[t], false, false);
      acc[t] = __builtin_amdgcn_wmma_f32_16x16x32_bf16(false, aHi, false, bLo, (short)0, acc[t], false, false);
      acc[t] = __builtin_amdgcn_wmma_f32_16x16x32_bf16(false, aLo, false, bHi, (short)0, acc[t], false, false);
    }
  }

  // D layout: VGPR r -> row (r + 8*half), col = lane&15.
#pragma unroll
  for (int t = 0; t < 4; ++t) {
    int col  = colBase + t * 16 + l15;
    float bv = bias ? bias[col] : 0.0f;
#pragma unroll
    for (int r = 0; r < 8; ++r) {
      int row = rowBase + r + half * 8;
      if (row < M) {
        float v = acc[t][r] + bv;
        if (relu) v = fmaxf(v, 0.0f);
        C[(size_t)row * Ncol + col] = v;
      }
    }
  }
}

// ---- per-row inverse norm (H = 256): one wave per row ---------------------
__global__ void k_rownorm(const float* __restrict__ X, float* __restrict__ rn, int N) {
  int row  = (int)((blockIdx.x * (size_t)blockDim.x + threadIdx.x) >> 5);
  int lane = threadIdx.x & 31;
  if (row >= N) return;
  const float* p = X + (size_t)row * 256 + lane * 8;
  float4 a = *(const float4*)p;
  float4 b = *(const float4*)(p + 4);
  float s = a.x*a.x + a.y*a.y + a.z*a.z + a.w*a.w
          + b.x*b.x + b.y*b.y + b.z*b.z + b.w*b.w;
  for (int off = 16; off; off >>= 1) s += __shfl_xor(s, off);
  if (lane == 0) rn[row] = 1.0f / fmaxf(sqrtf(s), 1e-12f);
}

// ---- per-edge cosine score + atomic segment-max: one wave per edge --------
__global__ void k_edge_score(const float* __restrict__ X, const float* __restrict__ rn,
                             const long long* __restrict__ ei, int E, int Etot,
                             const float* __restrict__ beta_p,
                             float* __restrict__ esc, unsigned* __restrict__ menc) {
  int e    = (int)((blockIdx.x * (size_t)blockDim.x + threadIdx.x) >> 5);
  int lane = threadIdx.x & 31;
  if (e >= Etot) return;
  int sIdx, dIdx;
  if (e < E) { sIdx = (int)ei[e]; dIdx = (int)ei[E + e]; }
  else       { sIdx = dIdx = e - E; }          // self-loop
  const float* xs = X + (size_t)sIdx * 256 + lane * 8;
  const float* xd = X + (size_t)dIdx * 256 + lane * 8;
  float4 a0 = *(const float4*)xs, a1 = *(const float4*)(xs + 4);
  float4 b0 = *(const float4*)xd, b1 = *(const float4*)(xd + 4);
  float acc = a0.x*b0.x + a0.y*b0.y + a0.z*b0.z + a0.w*b0.w
            + a1.x*b1.x + a1.y*b1.y + a1.z*b1.z + a1.w*b1.w;
  for (int off = 16; off; off >>= 1) acc += __shfl_xor(acc, off);
  if (lane == 0) {
    float beta = beta_p ? beta_p[0] : 1.0f;
    float sc   = beta * rn[sIdx] * rn[dIdx] * acc;
    esc[e] = sc;
    atomicMax(menc + dIdx, f2ord(sc));
  }
}

// ---- exp(score - max) + atomic segment-sum: one thread per edge -----------
__global__ void k_edge_exp(const long long* __restrict__ ei, int E, int Etot,
                           float* __restrict__ esc, const unsigned* __restrict__ menc,
                           float* __restrict__ ssum) {
  int e = blockIdx.x * blockDim.x + threadIdx.x;
  if (e >= Etot) return;
  int d   = (e < E) ? (int)ei[E + e] : (e - E);
  float v = expf(esc[e] - ord2f(menc[d]));
  esc[e]  = v;
  atomicAdd(ssum + d, v);
}

// ---- Y[dst] += (e/s[dst]) * X[src]: one wave per edge, atomic f32 adds ----
__global__ void k_edge_agg(const float* __restrict__ X, const long long* __restrict__ ei,
                           int E, int Etot, const float* __restrict__ esc,
                           const float* __restrict__ ssum, float* __restrict__ Y) {
  int e    = (int)((blockIdx.x * (size_t)blockDim.x + threadIdx.x) >> 5);
  int lane = threadIdx.x & 31;
  if (e >= Etot) return;
  int sIdx, dIdx;
  if (e < E) { sIdx = (int)ei[e]; dIdx = (int)ei[E + e]; }
  else       { sIdx = dIdx = e - E; }
  float w = esc[e] / ssum[dIdx];
  const float* xs = X + (size_t)sIdx * 256 + lane * 8;
  float*       yd = Y + (size_t)dIdx * 256 + lane * 8;
#pragma unroll
  for (int i = 0; i < 8; ++i) atomicAdd(yd + i, w * xs[i]);
}

// ---- log_softmax over 64 logits: one wave per row (2 vals / lane) ---------
__global__ void k_logsoftmax(const float* __restrict__ L, float* __restrict__ O, int N) {
  int row  = (int)((blockIdx.x * (size_t)blockDim.x + threadIdx.x) >> 5);
  int lane = threadIdx.x & 31;
  if (row >= N) return;
  const float* p = L + (size_t)row * 64;
  float a = p[lane * 2], b = p[lane * 2 + 1];
  float m = fmaxf(a, b);
  for (int off = 16; off; off >>= 1) m = fmaxf(m, __shfl_xor(m, off));
  float s = expf(a - m) + expf(b - m);
  for (int off = 16; off; off >>= 1) s += __shfl_xor(s, off);
  float lse = m + logf(s);
  float* o = O + (size_t)row * 64;
  o[lane * 2]     = a - lse;
  o[lane * 2 + 1] = b - lse;
}

// ---------------------------------------------------------------------------
extern "C" void kernel_launch(void* const* d_in, const int* in_sizes, int n_in,
                              void* d_out, int out_size, void* d_ws, size_t ws_size,
                              hipStream_t stream) {
  const float*     x     = (const float*)d_in[0];
  const long long* ei    = (const long long*)d_in[1];   // int64 [2, E]
  const float*     w1    = (const float*)d_in[2];
  const float*     b1    = (const float*)d_in[3];
  const float*     beta2 = (const float*)d_in[4];
  const float*     w2    = (const float*)d_in[5];
  const float*     b2    = (const float*)d_in[6];

  const int IN = 512, HID = 256, OUT = 64;
  const int N    = in_sizes[0] / IN;     // 50000
  const int E    = in_sizes[1] / 2;      // 800000
  const int Etot = E + N;                // + self loops

  // --- carve workspace ---
  char* ws = (char*)d_ws;
  auto carve = [&](size_t bytes) -> char* {
    char* p = ws;
    ws += (bytes + 255) & ~(size_t)255;
    return p;
  };
  __bf16* xhi  = (__bf16*)carve((size_t)N * IN * 2);
  __bf16* xlo  = (__bf16*)carve((size_t)N * IN * 2);
  __bf16* w1hi = (__bf16*)carve((size_t)HID * IN * 2);
  __bf16* w1lo = (__bf16*)carve((size_t)HID * IN * 2);
  __bf16* w2hi = (__bf16*)carve((size_t)OUT * HID * 2);
  __bf16* w2lo = (__bf16*)carve((size_t)OUT * HID * 2);
  float*    h      = (float*)carve((size_t)N * HID * 4);
  float*    rn     = (float*)carve((size_t)N * 4);
  unsigned* menc   = (unsigned*)carve((size_t)N * 4);
  float*    ssum   = (float*)carve((size_t)N * 4);
  float*    esc    = (float*)carve((size_t)Etot * 4);
  float*    logits = (float*)carve((size_t)N * OUT * 4);
  // z2's bf16 split reuses the x split region (x dead after GEMM1)
  __bf16* zhi = xhi;
  __bf16* zlo = xlo;

  float* z1  = (float*)d_out;                 // [N, 256]
  float* z2  = z1 + (size_t)N * HID;          // [N, 256]
  float* lsm = z2 + (size_t)N * HID;          // [N, 64]

  const int TPB = 256;
  auto cdiv = [](long long a, long long b) { return (int)((a + b - 1) / b); };

  // --- bf16 splits (weights + x) ---
  k_split_bf16<<<cdiv((long long)N * IN, TPB), TPB, 0, stream>>>(x, xhi, xlo, N * IN);
  k_split_bf16<<<cdiv((long long)HID * IN, TPB), TPB, 0, stream>>>(w1, w1hi, w1lo, HID * IN);
  k_split_bf16<<<cdiv((long long)OUT * HID, TPB), TPB, 0, stream>>>(w2, w2hi, w2lo, OUT * HID);

  // --- h = relu(x @ w1^T + b1) ---
  k_gemm_wmma<<<dim3(cdiv(N, 128), HID / 64), TPB, 0, stream>>>(
      xhi, xlo, w1hi, w1lo, b1, h, N, IN, HID, 1);

  // --- agnn_conv twice ---
  auto conv = [&](const float* X, float* Y, const float* betap) {
    k_rownorm<<<cdiv((long long)N * 32, TPB), TPB, 0, stream>>>(X, rn, N);
    hipMemsetAsync(menc, 0, (size_t)N * 4, stream);   // ord-encoded -inf floor
    hipMemsetAsync(ssum, 0, (size_t)N * 4, stream);
    hipMemsetAsync(Y, 0, (size_t)N * HID * 4, stream);
    k_edge_score<<<cdiv((long long)Etot * 32, TPB), TPB, 0, stream>>>(
        X, rn, ei, E, Etot, betap, esc, menc);
    k_edge_exp<<<cdiv(Etot, TPB), TPB, 0, stream>>>(ei, E, Etot, esc, menc, ssum);
    k_edge_agg<<<cdiv((long long)Etot * 32, TPB), TPB, 0, stream>>>(
        X, ei, E, Etot, esc, ssum, Y);
  };
  conv(h, z1, nullptr);   // beta = 1
  conv(z1, z2, beta2);    // learnable beta

  // --- logits = z2 @ w2^T + b2 ; log_softmax ---
  k_split_bf16<<<cdiv((long long)N * HID, TPB), TPB, 0, stream>>>(z2, zhi, zlo, N * HID);
  k_gemm_wmma<<<dim3(cdiv(N, 128), OUT / 64), TPB, 0, stream>>>(
      zhi, zlo, w2hi, w2lo, b2, logits, N, HID, OUT, 0);
  k_logsoftmax<<<cdiv((long long)N * 32, TPB), TPB, 0, stream>>>(logits, lsm, N);
}
